// GraphSAGE_3547642987106
// MI455X (gfx1250) — matrix operand order repaired and checked by reading, not compiled, not run on hardware
//
#include <hip/hip_runtime.h>

// ---------------------------------------------------------------------------
// GraphSAGE 2-layer forward for MI455X (gfx1250, wave32, WMMA).
//   layer: out = mean_agg(h)@W_l + h@W_r + b  ==  [mean | h](Nx256) @ [W_l;W_r](256xOC) + b
//   GEMM runs on v_wmma_f32_16x16x32_f16 (f16 in, f32 accumulate).
// ---------------------------------------------------------------------------

typedef __attribute__((ext_vector_type(16))) _Float16 v16h;
typedef __attribute__((ext_vector_type(8)))  _Float16 v8h;
typedef __attribute__((ext_vector_type(8)))  float    v8f;

#define DIM_F 128          // per-branch feature dim entering each layer
#define KTOT  256          // concat K = 2*DIM_F
#define KCHUNKS 8          // KTOT / 32 (WMMA K per instruction)

static inline size_t alignUp256(size_t x) { return (x + 255) & ~(size_t)255; }

// -------------------------------- zero fill --------------------------------
__global__ void zero_f32(float* __restrict__ p, long n) {
    long i = (long)blockIdx.x * blockDim.x + threadIdx.x;
    long stride = (long)gridDim.x * blockDim.x;
    for (; i < n; i += stride) p[i] = 0.0f;
}

// ------------------- weight pre-swizzle into WMMA B fragments --------------
// B fragment k order used here (contiguous, same as A; 32x16 KxN tile, wave32):
//   lane 0-15  -> column n = ct*16 + lane,   K = kk*32 + 0..15
//   lane 16-31 -> column n = ct*16 + lane-16,K = kk*32 + 16..31
// Stored per (frag, lane) as 16 contiguous f16 (32B) so GEMM loads one b128x2.
__global__ void prep_bfrags(const float* __restrict__ Wl,
                            const float* __restrict__ Wr,
                            int OC, int nct, _Float16* __restrict__ out) {
    int t = blockIdx.x * blockDim.x + threadIdx.x;
    int nfrag = KCHUNKS * nct;
    int frag = t >> 5;
    if (frag >= nfrag) return;
    int lane = t & 31;
    int kk = frag / nct, ct = frag % nct;
    int n  = ct * 16 + (lane & 15);
    int kb = kk * 32 + ((lane >> 4) << 4);       // +16 for upper half-wave
    _Float16* dst = out + ((size_t)frag * 32 + lane) * 16;
#pragma unroll
    for (int i = 0; i < 16; ++i) {
        int k = kb + i;
        float w = (k < DIM_F) ? Wl[(size_t)k * OC + n]
                              : Wr[(size_t)(k - DIM_F) * OC + n];
        dst[i] = (_Float16)w;
    }
}

// --------------------------- edge scatter-add ------------------------------
// One wave32 per edge: lane loads float4 of source row, 4 f32 atomics to dst.
__global__ void agg_scatter(const float* __restrict__ feat,
                            const long long* __restrict__ src,
                            const long long* __restrict__ dst,
                            float* __restrict__ agg,
                            float* __restrict__ deg,
                            int nEdges, int addDeg) {
    long gid = (long)blockIdx.x * blockDim.x + threadIdx.x;
    long e = gid >> 5;
    if (e >= nEdges) return;
    int lane = threadIdx.x & 31;
    long s = (long)src[e];
    long d = (long)dst[e];
    const float4 v = *(const float4*)(feat + s * DIM_F + lane * 4);
    float* a = agg + d * DIM_F + lane * 4;
    atomicAdd(a + 0, v.x);
    atomicAdd(a + 1, v.y);
    atomicAdd(a + 2, v.z);
    atomicAdd(a + 3, v.w);
    if (addDeg && lane == 0) atomicAdd(deg + d, 1.0f);
}

// ------------------------ fused mean + dual-GEMM ---------------------------
// Block = nct waves; handles 16 rows x OC cols. A tile (16 x 256 f16) staged
// in LDS: cols [0,128) = agg*inv_deg (mean branch), cols [128,256) = feat.
// Each wave owns one 16x16 output tile; 8x v_wmma_f32_16x16x32_f16 over K=256.
__global__ __launch_bounds__(256) void sage_gemm(
    const float* __restrict__ agg, const float* __restrict__ deg,
    const float* __restrict__ feat, const _Float16* __restrict__ Bfrags,
    const float* __restrict__ bias, float* __restrict__ out,
    int nNodes, int OC, int nct, int relu) {
    __shared__ _Float16 As[16 * KTOT];          // 8 KB

    const int rowBase = blockIdx.x * 16;
    const int nthr = nct * 32;

    // ---- stage A tile (mean | feat) as f16 ----
    for (int idx = threadIdx.x; idx < 16 * KTOT; idx += nthr) {
        int r = idx >> 8;                        // row in tile
        int c = idx & (KTOT - 1);                // col in concat K
        int row = rowBase + r;
        if (row >= nNodes) row = nNodes - 1;     // clamp (N % 16 == 0 normally)
        float v;
        if (c < DIM_F) {
            float inv = 1.0f / fmaxf(deg[row], 1.0f);
            v = agg[(size_t)row * DIM_F + c] * inv;
        } else {
            v = feat[(size_t)row * DIM_F + (c - DIM_F)];
        }
        As[idx] = (_Float16)v;
    }
    __syncthreads();

    const int wv   = threadIdx.x >> 5;           // column tile owned by wave
    const int lane = threadIdx.x & 31;

    v8f acc = {};
    // A fragment k order used here (contiguous, same as B; 16x32 MxK, wave32):
    //   lanes 0-15 : M=lane,    K = {0..15} of the 32-chunk
    //   lanes 16-31: M=lane-16, K = {16..31}
    const int m  = lane & 15;
    const int kl = (lane >> 4) << 4;             // 0 or 16
#pragma unroll
    for (int kk = 0; kk < KCHUNKS; ++kk) {
        int kb = kk * 32 + kl;
        const v8h lo = *(const v8h*)(&As[m * KTOT + kb]);        // K kb..kb+7
        const v8h hi = *(const v8h*)(&As[m * KTOT + kb + 8]);    // K kb+8..kb+15
        v16h a = __builtin_shufflevector(lo, hi,
                                         0, 1, 2, 3, 4, 5, 6, 7,
                                         8, 9, 10, 11, 12, 13, 14, 15);
        v16h b = *(const v16h*)(Bfrags + ((size_t)(kk * nct + wv) * 32 + lane) * 16);
        acc = __builtin_amdgcn_wmma_f32_16x16x32_f16(
            /*neg_a=*/false, a, /*neg_b=*/false, b,
            /*c_mod=*/(short)0, acc, /*reuse_a=*/false, /*reuse_b=*/false);
    }

    // ---- epilogue: bias (+ReLU), store. C/D layout: lane = col, VGPR = row.
    const int col = wv * 16 + (lane & 15);
    const int rb  = (lane >> 4) << 3;            // 0 or 8
    const float bv = bias[col];
#pragma unroll
    for (int r = 0; r < 8; ++r) {
        int row = rowBase + rb + r;
        if (row >= nNodes) continue;
        float v = acc[r] + bv;
        if (relu) v = fmaxf(v, 0.0f);
        out[(size_t)row * OC + col] = v;
    }
}

// ------------------------------ launcher -----------------------------------
extern "C" void kernel_launch(void* const* d_in, const int* in_sizes, int n_in,
                              void* d_out, int out_size, void* d_ws, size_t ws_size,
                              hipStream_t stream) {
    (void)n_in; (void)out_size; (void)ws_size;
    const float*     x   = (const float*)d_in[0];
    const long long* ei  = (const long long*)d_in[1];   // int64 (2, E)
    const float*     Wl1 = (const float*)d_in[2];
    const float*     Wr1 = (const float*)d_in[3];
    const float*     b1  = (const float*)d_in[4];
    const float*     Wl2 = (const float*)d_in[5];
    const float*     Wr2 = (const float*)d_in[6];
    const float*     b2  = (const float*)d_in[7];
    float*           out = (float*)d_out;

    const int N = in_sizes[0] / DIM_F;
    const int E = in_sizes[1] / 2;
    const long long* src = ei;
    const long long* dst = ei + E;

    // ---- workspace carve-out (~103 MB) ----
    char* ws = (char*)d_ws;
    size_t off = 0;
    float* deg = (float*)(ws + off); off = alignUp256(off + (size_t)N * 4);
    float* agg = (float*)(ws + off); off = alignUp256(off + (size_t)N * DIM_F * 4);
    float* h   = (float*)(ws + off); off = alignUp256(off + (size_t)N * DIM_F * 4);
    _Float16* Bf1 = (_Float16*)(ws + off); off = alignUp256(off + (size_t)KTOT * DIM_F * 2);
    _Float16* Bf2 = (_Float16*)(ws + off); off = alignUp256(off + (size_t)KTOT * 64 * 2);

    const int nBlocksRow = (N + 15) / 16;
    const int aggBlocks  = (int)(((long)E * 32 + 255) / 256);

    // layer 1
    zero_f32<<<2048, 256, 0, stream>>>(deg, N);
    zero_f32<<<4096, 256, 0, stream>>>(agg, (long)N * DIM_F);
    prep_bfrags<<<(KCHUNKS * 8 * 32 + 255) / 256, 256, 0, stream>>>(Wl1, Wr1, 128, 8, Bf1);
    prep_bfrags<<<(KCHUNKS * 4 * 32 + 255) / 256, 256, 0, stream>>>(Wl2, Wr2, 64, 4, Bf2);
    agg_scatter<<<aggBlocks, 256, 0, stream>>>(x, src, dst, agg, deg, E, /*addDeg=*/1);
    sage_gemm<<<nBlocksRow, 8 * 32, 0, stream>>>(agg, deg, x, Bf1, b1, h,
                                                 N, /*OC=*/128, /*nct=*/8, /*relu=*/1);
    // layer 2
    zero_f32<<<4096, 256, 0, stream>>>(agg, (long)N * DIM_F);
    agg_scatter<<<aggBlocks, 256, 0, stream>>>(h, src, dst, agg, deg, E, /*addDeg=*/0);
    sage_gemm<<<nBlocksRow, 4 * 32, 0, stream>>>(agg, deg, h, Bf2, b2, out,
                                                 N, /*OC=*/64, /*nct=*/4, /*relu=*/0);
}
